// SelfAttention_63453846831433
// MI455X (gfx1250) — matrix operand order repaired and checked by reading, not compiled, not run on hardware
//
#include <hip/hip_runtime.h>
#include <hip/hip_bf16.h>

// MI455X / gfx1250, wave32. All GEMMs via v_wmma_f32_16x16x32_bf16.
// Tile staging uses CDNA5 GLOBAL_LOAD_ASYNC_TO_LDS where the copy is
// address-contiguous (A tiles, V tiles); transposed stages use ds stores.

typedef __bf16 bf16_t;
typedef __bf16 v16bf __attribute__((ext_vector_type(16)));
typedef __bf16 v8bf  __attribute__((ext_vector_type(8)));
typedef float  v8f   __attribute__((ext_vector_type(8)));

#define DEVINL __device__ __forceinline__

DEVINL bf16_t f2bf(float f) {
  union { float f; unsigned u; } cf; cf.f = f;
  unsigned u = cf.u;
  u += 0x7FFFu + ((u >> 16) & 1u);           // round-to-nearest-even
  unsigned short s = (unsigned short)(u >> 16);
  union { unsigned short s; bf16_t b; } cb; cb.s = s;
  return cb.b;
}

// CDNA5 async memory->LDS copy (16 bytes per lane), ASYNCcnt-tracked.
DEVINL void async_g2l_b128(void* lds_dst, const void* gsrc) {
  unsigned loff = (unsigned)(unsigned long long)lds_dst;   // wave-relative LDS addr
  unsigned long long ga = (unsigned long long)gsrc;
  asm volatile("global_load_async_to_lds_b128 %0, %1, off"
               :: "v"(loff), "v"(ga)
               : "memory");
}
DEVINL void wait_async0() { asm volatile("s_wait_asynccnt 0" ::: "memory"); }

// ---------------------------------------------------------------------------
// f32 -> bf16 conversion (weights)
// ---------------------------------------------------------------------------
__global__ __launch_bounds__(256) void cvt_f32_bf16(const float* __restrict__ in,
                                                    bf16_t* __restrict__ out, int n) {
  int i = blockIdx.x * 256 + threadIdx.x;
  if (i < n) out[i] = f2bf(in[i]);
}

// ---------------------------------------------------------------------------
// GroupNorm (1 group): one block per batch sample; C=256, N=1024 -> 262144 el.
// ---------------------------------------------------------------------------
__global__ __launch_bounds__(256) void groupnorm_kernel(
    const float* __restrict__ xin, const float* __restrict__ w,
    const float* __restrict__ bi, bf16_t* __restrict__ outb, float eps) {
  __shared__ float s1[256];
  __shared__ float s2[256];
  const int b = blockIdx.x, t = threadIdx.x;
  const float* xb = xin + (long)b * 262144;
  float a = 0.f, a2 = 0.f;
  for (int idx = t; idx < 262144; idx += 256) {
    float v = xb[idx]; a += v; a2 += v * v;
  }
  s1[t] = a; s2[t] = a2;
  __syncthreads();
  for (int st = 128; st > 0; st >>= 1) {
    if (t < st) { s1[t] += s1[t + st]; s2[t] += s2[t + st]; }
    __syncthreads();
  }
  const float mu  = s1[0] * (1.f / 262144.f);
  const float var = s2[0] * (1.f / 262144.f) - mu * mu;
  const float inv = rsqrtf(var + eps);
  bf16_t* ob = outb + (long)b * 262144;
  for (int idx = t; idx < 262144; idx += 256) {
    int c = idx >> 10;
    ob[idx] = f2bf((xb[idx] - mu) * inv * w[c] + bi[c]);
  }
}

// ---------------------------------------------------------------------------
// Tiled bf16 WMMA GEMM: D = A[M,K] x B[K,N] (+ epilogue), batched over z.
// Block tile 128x128, 8 waves (4 along M x 2 along N), wave tile 32x64.
// EPI: 0 = store bf16
//      1 = f32: resid + gamma*(acc + bias[m])          (attn proj -> x2)
//      2 = bf16: silu(acc + bias[m])                   (FFN in)
//      3 = f32: resid + gamma*(acc + bias[m])          (FFN out -> d_out)
// ---------------------------------------------------------------------------
template <int EPI>
__global__ __launch_bounds__(256) void gemm_bf16_wmma(
    const bf16_t* __restrict__ A, const bf16_t* __restrict__ Bmat,
    void* __restrict__ D, const float* __restrict__ bias,
    const float* __restrict__ resid, const float* __restrict__ gamma,
    int M, int K, int Nn, long sBb, long sDb, long sRb) {
  __shared__ __align__(16) bf16_t Al[128 * 40];  // A tile [m][k], stride 40 (80B, 16B-mult)
  __shared__ __align__(16) bf16_t Bl[128 * 40];  // B tile transposed [n][k]
  const int t = threadIdx.x, lane = t & 31, wave = t >> 5;
  const int l15 = lane & 15, hf = lane >> 4;
  const int n0 = blockIdx.x * 128, m0 = blockIdx.y * 128, bz = blockIdx.z;
  const bf16_t* Bb = Bmat + (long)bz * sBb;
  const int wm = (wave & 3) * 32, wn = (wave >> 2) * 64;

  v8f acc[2][4];
#pragma unroll
  for (int i = 0; i < 2; i++)
#pragma unroll
    for (int j = 0; j < 4; j++)
#pragma unroll
      for (int r = 0; r < 8; r++) acc[i][j][r] = 0.f;

  for (int k0 = 0; k0 < K; k0 += 32) {
    __syncthreads();
    // Stage A tile 128x32: async global->LDS (16B per lane, no VGPR round-trip)
    for (int i = t; i < 512; i += 256) {
      int row = i >> 2, ck = (i & 3) * 8;
      async_g2l_b128(&Al[row * 40 + ck], &A[(long)(m0 + row) * K + k0 + ck]);
    }
    // Stage B tile 32x128, transposed into [n][k] (coalesced global reads over n)
    for (int i = t; i < 4096; i += 256) {
      int kk = i >> 7, nn = i & 127;
      Bl[nn * 40 + kk] = Bb[(long)(k0 + kk) * Nn + n0 + nn];
    }
    if (k0 + 32 < K) {  // hint next tiles into cache (global_prefetch_b8)
      __builtin_prefetch(&Bb[(long)(k0 + 32 + (t >> 3)) * Nn + n0 + (t & 7) * 16], 0, 1);
    }
    wait_async0();       // async A-tile landed in LDS
    __syncthreads();

    // A fragment: lane l15 = row m; lanes 0-15 hold K {0..7,16..23}, 16-31 hold {8..15,24..31}
    v16bf af[2], bfr[4];
#pragma unroll
    for (int mi = 0; mi < 2; mi++) {
      const bf16_t* p = &Al[(wm + mi * 16 + l15) * 40 + 8 * hf];
      ((v8bf*)&af[mi])[0] = *(const v8bf*)p;
      ((v8bf*)&af[mi])[1] = *(const v8bf*)(p + 16);
    }
    // B fragment: lane l15 = col n; lanes 0-15 hold K {0..7,16..23}, 16-31 hold {8..15,24..31}
#pragma unroll
    for (int ni = 0; ni < 4; ni++) {
      const bf16_t* p = &Bl[(wn + ni * 16 + l15) * 40 + 8 * hf];
      ((v8bf*)&bfr[ni])[0] = *(const v8bf*)p;
      ((v8bf*)&bfr[ni])[1] = *(const v8bf*)(p + 16);
    }
#pragma unroll
    for (int mi = 0; mi < 2; mi++)
#pragma unroll
      for (int ni = 0; ni < 4; ni++)
        acc[mi][ni] = __builtin_amdgcn_wmma_f32_16x16x32_bf16(
            false, af[mi], false, bfr[ni], (short)0, acc[mi][ni], false, false);
  }

  // Epilogue. C/D layout: VGPR r -> M = r + 8*(lane>=16); N = lane&15.
  const float g = (EPI == 1 || EPI == 3) ? gamma[0] : 0.f;
#pragma unroll
  for (int mi = 0; mi < 2; mi++)
#pragma unroll
    for (int ni = 0; ni < 4; ni++)
#pragma unroll
      for (int r = 0; r < 8; r++) {
        int gm = m0 + wm + mi * 16 + 8 * hf + r;
        int gn = n0 + wn + ni * 16 + l15;
        float v = acc[mi][ni][r];
        long di = (long)gm * Nn + gn;
        if (EPI == 0) {
          ((bf16_t*)D)[(long)bz * sDb + di] = f2bf(v);
        } else if (EPI == 1 || EPI == 3) {
          float o = resid[(long)bz * sRb + di] + g * (v + bias[gm]);
          ((float*)D)[(long)bz * sDb + di] = o;
        } else {  // EPI == 2: bias + SiLU, bf16 out
          float o = v + bias[gm];
          o = o / (1.f + __expf(-o));
          ((bf16_t*)D)[(long)bz * sDb + di] = f2bf(o);
        }
      }
}

// ---------------------------------------------------------------------------
// Flash attention: per (i-block of 128, head, batch). dk=64, N=1024, NH=4.
// qkv is bf16 [B, 768, 1024] (q rows 0..255, k 256..511, v 512..767).
// Online softmax over 64-wide j chunks; S and O GEMMs via WMMA.
// ---------------------------------------------------------------------------
__global__ __launch_bounds__(256) void flash_attn_wmma(
    const bf16_t* __restrict__ qkv, bf16_t* __restrict__ outm) {
  __shared__ __align__(16) bf16_t Qt[128 * 72];  // Qt[i][d]  (A for S)
  __shared__ __align__(16) bf16_t Kl[64 * 72];   // Kl[j][d]  (B for S, [n][k])
  __shared__ __align__(16) bf16_t Vl[64 * 72];   // Vl[d][j]  (B for O, [n][k])
  __shared__ __align__(16) bf16_t Pl[128 * 72];  // P [i][j]  (A for O), wave-private rows
  const int t = threadIdx.x, lane = t & 31, wave = t >> 5;
  const int l15 = lane & 15, hf = lane >> 4;
  const int i0 = blockIdx.x * 128, h = blockIdx.y, bz = blockIdx.z;
  const bf16_t* base = qkv + (long)bz * 768 * 1024;
  const bf16_t* Qp = base + (h * 64) * 1024;
  const bf16_t* Kp = base + (256 + h * 64) * 1024;
  const bf16_t* Vp = base + (512 + h * 64) * 1024;

  // Stage Q transposed: Qt[i][d] = Q[d][i0+i]  (coalesced over i)
  for (int idx = t; idx < 128 * 64; idx += 256) {
    int d = idx >> 7, i = idx & 127;
    Qt[i * 72 + d] = Qp[d * 1024 + i0 + i];
  }

  v8f o[4];
  float rmax[8], rsum[8];
#pragma unroll
  for (int dt = 0; dt < 4; dt++)
#pragma unroll
    for (int r = 0; r < 8; r++) o[dt][r] = 0.f;
#pragma unroll
  for (int r = 0; r < 8; r++) { rmax[r] = -3.0e38f; rsum[r] = 0.f; }

  for (int j0 = 0; j0 < 1024; j0 += 64) {
    __syncthreads();
    // V tile is a direct row copy -> async global->LDS (64 rows x 128B)
    for (int i = t; i < 512; i += 256) {
      int d = i >> 3, cj = (i & 7) * 8;
      async_g2l_b128(&Vl[d * 72 + cj], &Vp[d * 1024 + j0 + cj]);
    }
    // K tile is transposed (Kl[j][d] = K[d][j]) -> ds stores
    for (int idx = t; idx < 64 * 64; idx += 256) {
      int d = idx >> 6, j = idx & 63;
      Kl[j * 72 + d] = Kp[d * 1024 + j0 + j];
    }
    wait_async0();
    __syncthreads();

    // S tile: 16 query rows per wave x 64 j cols
    v8f s[4];
#pragma unroll
    for (int jt = 0; jt < 4; jt++)
#pragma unroll
      for (int r = 0; r < 8; r++) s[jt][r] = 0.f;
#pragma unroll
    for (int ks = 0; ks < 64; ks += 32) {
      v16bf aq;
      const bf16_t* p = &Qt[(wave * 16 + l15) * 72 + ks + 8 * hf];
      ((v8bf*)&aq)[0] = *(const v8bf*)p;
      ((v8bf*)&aq)[1] = *(const v8bf*)(p + 16);
#pragma unroll
      for (int jt = 0; jt < 4; jt++) {
        v16bf bk;
        const bf16_t* q = &Kl[(jt * 16 + l15) * 72 + ks + 8 * hf];
        ((v8bf*)&bk)[0] = *(const v8bf*)q;
        ((v8bf*)&bk)[1] = *(const v8bf*)(q + 16);
        s[jt] = __builtin_amdgcn_wmma_f32_16x16x32_bf16(
            false, aq, false, bk, (short)0, s[jt], false, false);
      }
    }

    // scale 1/sqrt(dk) + online softmax (row spans 16 lanes of one half)
#pragma unroll
    for (int jt = 0; jt < 4; jt++)
#pragma unroll
      for (int r = 0; r < 8; r++) s[jt][r] *= 0.125f;
    float corr[8];
#pragma unroll
    for (int r = 0; r < 8; r++) {
      float mx = s[0][r];
#pragma unroll
      for (int jt = 1; jt < 4; jt++) mx = fmaxf(mx, s[jt][r]);
#pragma unroll
      for (int m = 1; m < 16; m <<= 1) mx = fmaxf(mx, __shfl_xor(mx, m, 32));
      float nm = fmaxf(rmax[r], mx);
      float c = __expf(rmax[r] - nm);
      float ls = 0.f;
#pragma unroll
      for (int jt = 0; jt < 4; jt++) {
        float pv = __expf(s[jt][r] - nm);
        s[jt][r] = pv; ls += pv;
      }
#pragma unroll
      for (int m = 1; m < 16; m <<= 1) ls += __shfl_xor(ls, m, 32);
      rmax[r] = nm; rsum[r] = rsum[r] * c + ls; corr[r] = c;
    }
#pragma unroll
    for (int dt = 0; dt < 4; dt++)
#pragma unroll
      for (int r = 0; r < 8; r++) o[dt][r] *= corr[r];

    // P (C-fragment layout) -> LDS rows owned by this wave -> A-fragments
#pragma unroll
    for (int jt = 0; jt < 4; jt++)
#pragma unroll
      for (int r = 0; r < 8; r++)
        Pl[(wave * 16 + 8 * hf + r) * 72 + jt * 16 + l15] = f2bf(s[jt][r]);

    // O += P x V^T   (B[k=j][n=d] staged as Vl[d][j])
#pragma unroll
    for (int ks = 0; ks < 64; ks += 32) {
      v16bf ap;
      const bf16_t* p = &Pl[(wave * 16 + l15) * 72 + ks + 8 * hf];
      ((v8bf*)&ap)[0] = *(const v8bf*)p;
      ((v8bf*)&ap)[1] = *(const v8bf*)(p + 16);
#pragma unroll
      for (int dt = 0; dt < 4; dt++) {
        v16bf bv;
        const bf16_t* q = &Vl[(dt * 16 + l15) * 72 + ks + 8 * hf];
        ((v8bf*)&bv)[0] = *(const v8bf*)q;
        ((v8bf*)&bv)[1] = *(const v8bf*)(q + 16);
        o[dt] = __builtin_amdgcn_wmma_f32_16x16x32_bf16(
            false, ap, false, bv, (short)0, o[dt], false, false);
      }
    }
  }

  // Normalize and store transposed: out[c = h*64+d][n = i]
  bf16_t* Ob = outm + (long)bz * 256 * 1024 + (h * 64) * 1024;
#pragma unroll
  for (int dt = 0; dt < 4; dt++)
#pragma unroll
    for (int r = 0; r < 8; r++) {
      int d = dt * 16 + l15;
      int i = i0 + wave * 16 + 8 * hf + r;
      Ob[d * 1024 + i] = f2bf(o[dt][r] / rsum[r]);
    }
}

// ---------------------------------------------------------------------------
// Host-side orchestration
// ---------------------------------------------------------------------------
extern "C" void kernel_launch(void* const* d_in, const int* in_sizes, int n_in,
                              void* d_out, int out_size, void* d_ws, size_t ws_size,
                              hipStream_t stream) {
  (void)in_sizes; (void)n_in; (void)out_size; (void)ws_size;
  const float* x      = (const float*)d_in[0];
  const float* qkv_w  = (const float*)d_in[1];
  const float* uh_w   = (const float*)d_in[2];
  const float* uh_b   = (const float*)d_in[3];
  const float* n1_w   = (const float*)d_in[4];
  const float* n1_b   = (const float*)d_in[5];
  const float* n2_w   = (const float*)d_in[6];
  const float* n2_b   = (const float*)d_in[7];
  const float* f1_w   = (const float*)d_in[8];
  const float* f1_b   = (const float*)d_in[9];
  const float* f2_w   = (const float*)d_in[10];
  const float* f2_b   = (const float*)d_in[11];
  const float* g_attn = (const float*)d_in[12];
  const float* g_ffn  = (const float*)d_in[13];

  char* ws = (char*)d_ws;
  bf16_t* w_qkv = (bf16_t*)(ws + 0);         // 768*256*2  = 384 KB
  bf16_t* w_uh  = (bf16_t*)(ws + 393216);    // 256*256*2  = 128 KB
  bf16_t* w_f1  = (bf16_t*)(ws + 524288);    // 1024*256*2 = 512 KB
  bf16_t* w_f2  = (bf16_t*)(ws + 1048576);   // 256*1024*2 = 512 KB
  bf16_t* xn    = (bf16_t*)(ws + 1572864);   // [B,256,1024] bf16, 4 MB (reused GN1/GN2)
  bf16_t* qkvb  = (bf16_t*)(ws + 5767168);   // [B,768,1024] bf16, 12 MB
  bf16_t* h1    = (bf16_t*)(ws + 5767168);   // [B,1024,1024] bf16, 16 MB (aliases qkv+attn, dead by then)
  bf16_t* attn  = (bf16_t*)(ws + 18350080);  // [B,256,1024] bf16, 4 MB
  float*  x2    = (float*)(ws + 22544384);   // [B,256,1024] f32, 8 MB  (total ~29.5 MB)

  dim3 blk(256);
  cvt_f32_bf16<<<dim3(768), blk, 0, stream>>>(qkv_w, w_qkv, 196608);
  cvt_f32_bf16<<<dim3(256), blk, 0, stream>>>(uh_w, w_uh, 65536);
  cvt_f32_bf16<<<dim3(1024), blk, 0, stream>>>(f1_w, w_f1, 262144);
  cvt_f32_bf16<<<dim3(1024), blk, 0, stream>>>(f2_w, w_f2, 262144);

  // xn = GN1(x)
  groupnorm_kernel<<<dim3(8), blk, 0, stream>>>(x, n1_w, n1_b, xn, 1e-5f);
  // qkv = W_qkv @ xn   [768x256 x 256x1024] per batch
  gemm_bf16_wmma<0><<<dim3(8, 6, 8), blk, 0, stream>>>(
      w_qkv, xn, qkvb, nullptr, nullptr, nullptr, 768, 256, 1024, 262144L, 786432L, 0L);
  // attention (fused softmax)
  flash_attn_wmma<<<dim3(8, 4, 8), blk, 0, stream>>>(qkvb, attn);
  // x2 = x + gamma_attn * (W_uh @ attn + uh_b)
  gemm_bf16_wmma<1><<<dim3(8, 2, 8), blk, 0, stream>>>(
      w_uh, attn, x2, uh_b, x, g_attn, 256, 256, 1024, 262144L, 262144L, 262144L);
  // xn = GN2(x2)
  groupnorm_kernel<<<dim3(8), blk, 0, stream>>>(x2, n2_w, n2_b, xn, 1e-5f);
  // h1 = silu(W_f1 @ xn + f1_b)
  gemm_bf16_wmma<2><<<dim3(8, 8, 8), blk, 0, stream>>>(
      w_f1, xn, h1, f1_b, nullptr, nullptr, 1024, 256, 1024, 262144L, 1048576L, 0L);
  // out = x2 + gamma_ffn * (W_f2 @ h1 + f2_b)
  gemm_bf16_wmma<3><<<dim3(8, 2, 8), blk, 0, stream>>>(
      w_f2, h1, (float*)d_out, f2_b, x2, g_ffn, 256, 1024, 1024, 1048576L, 262144L, 262144L);
}
